// RQBottleneckTransformer_41223096107730
// MI455X (gfx1250) — compile-verified, matrix operand order checked by reading, not run on hardware
//
#include <hip/hip_runtime.h>
#include <hip/hip_bf16.h>
#include <math.h>

// ---------------- constants (RQBottleneckTransformer) ----------------
#define B_   16
#define T_   1500
#define S_   750      // T / DS
#define D_   512
#define H_   8
#define DH_  64
#define L_   4
#define V_   4097
#define C_   64
#define FF_  2048

typedef _Float16 f16;
typedef _Float16 v16h __attribute__((ext_vector_type(16)));
typedef float    v8f  __attribute__((ext_vector_type(8)));

#define AS1 __attribute__((address_space(1)))
#define AS3 __attribute__((address_space(3)))

// ---- CDNA5 async / tensor data-mover availability probes ----
#ifdef __has_builtin
#  if __has_builtin(__builtin_amdgcn_tensor_load_to_lds) && \
      __has_builtin(__builtin_amdgcn_s_wait_tensorcnt)
#    define USE_TDM 1
#  endif
#  if __has_builtin(__builtin_amdgcn_global_load_async_to_lds_b128) && \
      __has_builtin(__builtin_amdgcn_s_wait_asynccnt)
#    define USE_ASYNC 1
#  endif
#endif
#if defined(USE_TDM) || defined(USE_ASYNC)
#  define STAGE_A_LDS 1
#endif

union F16Frag {
    v16h h;
    uint4 u[2];
    f16  e[16];
};

__device__ inline v8f wmma16x16x32(v16h a, v16h b, v8f c) {
    // 8 args: (neg_a, A, neg_b, B, c_mod, C, reuse_a, reuse_b)
    return __builtin_amdgcn_wmma_f32_16x16x32_f16(false, a, false, b, (short)0, c,
                                                  false, false);
}

__device__ inline float gelu_exact(float v) {
    return 0.5f * v * (1.0f + erff(v * 0.70710678118f));
}

__device__ inline unsigned lds_off_bytes(void* p) {
    return (unsigned)(unsigned long long)(AS3 void*)p;
}

#ifdef USE_TDM
typedef unsigned int uint32x4 __attribute__((ext_vector_type(4)));
typedef int          int32x8  __attribute__((ext_vector_type(8)));
typedef int          int32x4  __attribute__((ext_vector_type(4)));

// TDM load of a 128-row x 32-elem f16 tile from a row-major [H x W] f16
// tensor into LDS, padding +4 DWORDs every 16 DWORDs (row stride 40 halves).
__device__ inline void tdm_load_a_tile(const void* gsrc, unsigned ldsOff,
                                       unsigned tensorW, unsigned tensorH,
                                       unsigned strideW) {
    const unsigned long long ga = (unsigned long long)(size_t)gsrc;
    uint32x4 g0;
    g0.x = 1u;                                   // count=1 valid user descriptor
    g0.y = ldsOff;                               // lds_addr
    g0.z = (unsigned)(ga & 0xffffffffu);         // global_addr[31:0]
    g0.w = (unsigned)((ga >> 32) & 0x1ffffffu) | (2u << 30);  // ga[56:32] | type=2
    int32x8 g1;
    // data_size=1 (2B) | pad_enable | pad_interval=3 (16 DW) | pad_amount=3 (4 DW)
    g1[0] = (int)((1u << 16) | (1u << 20) | (3u << 22) | (3u << 25));
    g1[1] = (int)((tensorW & 0xffffu) << 16);                     // dim0[15:0]
    g1[2] = (int)(((tensorW >> 16) & 0xffffu) | ((tensorH & 0xffffu) << 16));
    g1[3] = (int)(((tensorH >> 16) & 0xffffu) | (32u << 16));     // tile_dim0=32
    g1[4] = 128;                                                  // tile_dim1=128
    g1[5] = (int)strideW;                                         // dim0_stride
    g1[6] = 0;
    g1[7] = 0;
    int32x4 z4 = {0, 0, 0, 0};
#if defined(__clang_major__) && (__clang_major__ >= 23)
    int32x8 z8 = {0, 0, 0, 0, 0, 0, 0, 0};
    __builtin_amdgcn_tensor_load_to_lds(g0, g1, z4, z4, z8, 0);
#else
    __builtin_amdgcn_tensor_load_to_lds(g0, g1, z4, z4, 0);
#endif
}
#endif  // USE_TDM

// =====================================================================
// Generic WMMA GEMM:  C[M,N] = A[M,K](f16,row-major) @ Bm[K,N](f16,row-major)
// macro tile 128(M) x 64(N), 256 threads = 8 waves, each wave a 16x64 strip.
// A tile staged into LDS via TDM (or async global->LDS copies); B tile staged
// transposed in LDS so fragments are 16B ds_load_b128s.
// EPI: 0 = f32 out (acc+bias)        1 = f16 out gelu(acc+bias)
//      2 = f32 out (acc+bias+resid)  3 = f16 out (acc+bias)
// =====================================================================
template <int EPI>
__global__ __launch_bounds__(256) void gemm_wmma_f16(
    const f16* __restrict__ A, const f16* __restrict__ Bm,
    const float* __restrict__ bias, const float* __restrict__ resid,
    float* __restrict__ Cf, f16* __restrict__ Ch, int M, int N, int K)
{
    __shared__ f16 Bt[64][40];   // B^T tile [n][k], padded rows (80B stride)
#ifdef STAGE_A_LDS
    __shared__ f16 At[128][40];  // A tile [m][k], padded rows (80B stride)
#endif

    const int tid  = threadIdx.x;
    const int lane = tid & 31;
    const int wave = tid >> 5;
    const int g    = lane >> 4;     // lane group (0/1)
    const int ln16 = lane & 15;
    const int mBase0 = blockIdx.x * 128;
    const int mBase  = mBase0 + wave * 16;
    const int nBase  = blockIdx.y * 64;

    v8f acc[4] = {v8f{}, v8f{}, v8f{}, v8f{}};

    int rowA = mBase + ln16;
    if (rowA >= M) rowA = M - 1;
    const f16* arow = A + (size_t)rowA * K;

#ifdef USE_TDM
    const unsigned atOff = lds_off_bytes(&At[0][0]);
#endif

    for (int k0 = 0; k0 < K; k0 += 32) {
        // ---- stage A[128][32] into LDS (TDM / async / none) ----
#ifdef USE_TDM
        if (wave == 0) {
            tdm_load_a_tile(A + (size_t)mBase0 * K + k0, atOff,
                            (unsigned)K, (unsigned)M, (unsigned)K);
            __builtin_amdgcn_s_wait_tensorcnt(0);
        }
#elif defined(USE_ASYNC)
        {
#pragma unroll
            for (int j = 0; j < 2; ++j) {
                const int cid = tid + j * 256;   // 0..511 16B chunks
                const int rr  = cid >> 2;        // tile row 0..127
                const int qq  = cid & 3;         // 16B chunk within row
                int grow = mBase0 + rr;
                if (grow >= M) grow = M - 1;
                __builtin_amdgcn_global_load_async_to_lds_b128(
                    (AS1 void*)(A + (size_t)grow * K + k0 + qq * 8),
                    (AS3 void*)&At[rr][qq * 8], 0, 0);
            }
            __builtin_amdgcn_s_wait_asynccnt(0);
        }
#endif

        // ---- stage B[32][64] into Bt[n][k] (transposed) ----
        {
            const int kk = tid >> 3;          // 0..31
            const int n0 = (tid & 7) * 8;     // 0..56
            const f16* src = Bm + (size_t)(k0 + kk) * N + nBase + n0;
            uint4 pkt = *(const uint4*)src;
            const f16* pv = (const f16*)&pkt;
#pragma unroll
            for (int j = 0; j < 8; ++j) Bt[n0 + j][kk] = pv[j];
        }
        __syncthreads();

        // ---- A fragment: two contiguous 16B runs per lane ----
        F16Frag af;
#ifdef STAGE_A_LDS
        {
            const f16* ap = &At[wave * 16 + ln16][8 * g];
            af.u[0] = *(const uint4*)ap;
            af.u[1] = *(const uint4*)(ap + 16);
        }
#else
        af.u[0] = *(const uint4*)(arow + k0 + 8 * g);
        af.u[1] = *(const uint4*)(arow + k0 + 16 + 8 * g);
#endif

        if (k0 + 32 < K) {
            __builtin_prefetch(arow + k0 + 32, 0, 1);
            __builtin_prefetch(Bm + (size_t)(k0 + 32 + (tid >> 3)) * N + nBase, 0, 1);
        }

#pragma unroll
        for (int nt = 0; nt < 4; ++nt) {
            F16Frag bf;
            const f16* bp = &Bt[nt * 16 + ln16][8 * g];
            bf.u[0] = *(const uint4*)bp;
            bf.u[1] = *(const uint4*)(bp + 16);
            acc[nt] = wmma16x16x32(af.h, bf.h, acc[nt]);
        }
        __syncthreads();
    }

    // ---- epilogue ----
#pragma unroll
    for (int nt = 0; nt < 4; ++nt) {
        const int col = nBase + nt * 16 + ln16;
        const float bs = bias ? bias[col] : 0.0f;
#pragma unroll
        for (int r = 0; r < 8; ++r) {
            const int row = mBase + r + 8 * g;
            if (row >= M) continue;
            float v = acc[nt][r] + bs;
            if (EPI == 1) v = gelu_exact(v);
            if (EPI == 2) v += resid[(size_t)row * N + col];
            if (EPI == 0 || EPI == 2) Cf[(size_t)row * N + col] = v;
            else                      Ch[(size_t)row * N + col] = (f16)v;
        }
    }
}

// =====================================================================
// VQ distance argmin GEMM:  dist(row,v) = -2 * (z @ E^T) + ||E_v||^2
// =====================================================================
__device__ inline unsigned long long pack_dist(float d, int idx) {
    unsigned u = __float_as_uint(d);
    u = (u & 0x80000000u) ? ~u : (u | 0x80000000u);
    return ((unsigned long long)u << 32) | (unsigned)idx;
}

__global__ __launch_bounds__(256) void vq_argmin_gemm(
    const f16* __restrict__ Az, const f16* __restrict__ Et,
    const float* __restrict__ e2, unsigned long long* __restrict__ minbuf, int M)
{
    __shared__ f16 Bt[64][40];
    const int tid  = threadIdx.x;
    const int lane = tid & 31;
    const int wave = tid >> 5;
    const int g    = lane >> 4;
    const int ln16 = lane & 15;
    const int mBase = blockIdx.x * 128 + wave * 16;
    const int nBase = blockIdx.y * 64;

    v8f acc[4] = {v8f{}, v8f{}, v8f{}, v8f{}};

    int rowA = mBase + ln16;
    if (rowA >= M) rowA = M - 1;
    const f16* arow = Az + (size_t)rowA * C_;

    for (int k0 = 0; k0 < C_; k0 += 32) {
        {   // stage (scalar loads: N=4097 tail is not 16B aligned)
            const int kk = tid >> 3;
            const int n0 = (tid & 7) * 8;
#pragma unroll
            for (int j = 0; j < 8; ++j) {
                int n = nBase + n0 + j;
                if (n >= V_) n = V_ - 1;
                Bt[n0 + j][kk] = Et[(size_t)(k0 + kk) * V_ + n];
            }
        }
        __syncthreads();
        F16Frag af;
        af.u[0] = *(const uint4*)(arow + k0 + 8 * g);
        af.u[1] = *(const uint4*)(arow + k0 + 16 + 8 * g);
#pragma unroll
        for (int nt = 0; nt < 4; ++nt) {
            F16Frag bf;
            const f16* bp = &Bt[nt * 16 + ln16][8 * g];
            bf.u[0] = *(const uint4*)bp;
            bf.u[1] = *(const uint4*)(bp + 16);
            acc[nt] = wmma16x16x32(af.h, bf.h, acc[nt]);
        }
        __syncthreads();
    }

#pragma unroll
    for (int nt = 0; nt < 4; ++nt) {
        const int col = nBase + nt * 16 + ln16;
#pragma unroll
        for (int r = 0; r < 8; ++r) {
            float dv = (col < V_) ? (-2.0f * acc[nt][r] + e2[col]) : 3.4e38f;
            int   ci = (col < V_) ? col : 0;
#pragma unroll
            for (int mm = 1; mm < 16; mm <<= 1) {
                float od = __shfl_xor(dv, mm, 32);
                int   oc = __shfl_xor(ci, mm, 32);
                if (od < dv) { dv = od; ci = oc; }
            }
            if (ln16 == 0) {
                const int row = mBase + r + 8 * g;
                if (row < M) atomicMin(&minbuf[row], pack_dist(dv, ci));
            }
        }
    }
}

// =====================================================================
// Flash attention, one wave per (b, h, 16-row q tile). QK_SCALE == 1.0.
// =====================================================================
__global__ __launch_bounds__(32) void flash_attention(
    const f16* __restrict__ q16, const f16* __restrict__ k16,
    const f16* __restrict__ v16, f16* __restrict__ o16, float scale)
{
    const int lane = threadIdx.x & 31;
    const int g    = lane >> 4;
    const int ln16 = lane & 15;
    const int qbase = blockIdx.x * 16;
    const int h = blockIdx.y;
    const int b = blockIdx.z;
    const int hd0 = h * DH_;

    __shared__ f16 P[16][40];

    F16Frag qf[2];
    {
        int row = qbase + ln16; if (row >= T_) row = T_ - 1;
        const f16* qp = q16 + ((size_t)(b * T_ + row) * D_ + hd0);
#pragma unroll
        for (int f = 0; f < 2; ++f) {
            qf[f].u[0] = *(const uint4*)(qp + f * 32 + 8 * g);
            qf[f].u[1] = *(const uint4*)(qp + f * 32 + 16 + 8 * g);
        }
    }

    float mrun[8], lrun[8];
    v8f o[4] = {v8f{}, v8f{}, v8f{}, v8f{}};
#pragma unroll
    for (int r = 0; r < 8; ++r) { mrun[r] = -1e30f; lrun[r] = 0.0f; }

    for (int key0 = 0; key0 < T_; key0 += 32) {
        v8f s0 = v8f{}, s1 = v8f{};
        int keyA = key0 + ln16;      if (keyA >= T_) keyA = T_ - 1;
        int keyB = key0 + 16 + ln16; if (keyB >= T_) keyB = T_ - 1;
#pragma unroll
        for (int f = 0; f < 2; ++f) {
            const f16* kpA = k16 + ((size_t)(b * T_ + keyA) * D_ + hd0 + f * 32);
            F16Frag kfA;
            kfA.u[0] = *(const uint4*)(kpA + 8 * g);
            kfA.u[1] = *(const uint4*)(kpA + 16 + 8 * g);
            s0 = wmma16x16x32(qf[f].h, kfA.h, s0);
            const f16* kpB = k16 + ((size_t)(b * T_ + keyB) * D_ + hd0 + f * 32);
            F16Frag kfB;
            kfB.u[0] = *(const uint4*)(kpB + 8 * g);
            kfB.u[1] = *(const uint4*)(kpB + 16 + 8 * g);
            s1 = wmma16x16x32(qf[f].h, kfB.h, s1);
        }

        const bool omA = (key0 + ln16) >= T_;
        const bool omB = (key0 + 16 + ln16) >= T_;

#pragma unroll
        for (int r = 0; r < 8; ++r) {
            float a0 = omA ? -1e30f : s0[r] * scale;
            float a1 = omB ? -1e30f : s1[r] * scale;
            float mx = fmaxf(a0, a1);
#pragma unroll
            for (int mm = 1; mm < 16; mm <<= 1) mx = fmaxf(mx, __shfl_xor(mx, mm, 32));
            const float mnew  = fmaxf(mrun[r], mx);
            const float alpha = __expf(mrun[r] - mnew);
            const float p0 = __expf(a0 - mnew);
            const float p1 = __expf(a1 - mnew);
            float ps = p0 + p1;
#pragma unroll
            for (int mm = 1; mm < 16; mm <<= 1) ps += __shfl_xor(ps, mm, 32);
            lrun[r] = lrun[r] * alpha + ps;
            mrun[r] = mnew;
#pragma unroll
            for (int nt = 0; nt < 4; ++nt) o[nt][r] *= alpha;
            P[r + 8 * g][ln16]      = (f16)p0;
            P[r + 8 * g][16 + ln16] = (f16)p1;
        }
        __syncthreads();

        F16Frag pf;  // P as A-fragment (16 rows x 32 keys)
        pf.u[0] = *(const uint4*)&P[ln16][8 * g];
        pf.u[1] = *(const uint4*)&P[ln16][16 + 8 * g];

#pragma unroll
        for (int nt = 0; nt < 4; ++nt) {
            F16Frag vf;
#pragma unroll
            for (int e = 0; e < 16; ++e) {
                const int kk = (e < 8 ? e : e + 8) + 8 * g;
                int key = key0 + kk; if (key >= T_) key = T_ - 1;
                vf.e[e] = v16[(size_t)(b * T_ + key) * D_ + hd0 + nt * 16 + ln16];
            }
            o[nt] = wmma16x16x32(pf.h, vf.h, o[nt]);
        }
        __syncthreads();
    }

#pragma unroll
    for (int r = 0; r < 8; ++r) {
        const int row = qbase + r + 8 * g;
        if (row >= T_) continue;
        const float inv = 1.0f / lrun[r];
#pragma unroll
        for (int nt = 0; nt < 4; ++nt)
            o16[(size_t)(b * T_ + row) * D_ + hd0 + nt * 16 + ln16] =
                (f16)(o[nt][r] * inv);
    }
}

// ---------------- LayerNorm over D=512, one wave per row ----------------
template <bool OUT16>
__global__ __launch_bounds__(256) void layernorm_k(
    const float* __restrict__ x, const float* __restrict__ gw,
    const float* __restrict__ bw, f16* __restrict__ y16,
    float* __restrict__ y32, int rows)
{
    const int wave = threadIdx.x >> 5, lane = threadIdx.x & 31;
    const int row = blockIdx.x * 8 + wave;
    if (row >= rows) return;
    const float* xr = x + (size_t)row * D_;
    float v[16];
    float s = 0.0f;
#pragma unroll
    for (int j = 0; j < 16; ++j) { v[j] = xr[lane + 32 * j]; s += v[j]; }
#pragma unroll
    for (int mm = 1; mm < 32; mm <<= 1) s += __shfl_xor(s, mm, 32);
    const float mean = s * (1.0f / 512.0f);
    float vs = 0.0f;
#pragma unroll
    for (int j = 0; j < 16; ++j) { float d = v[j] - mean; vs += d * d; }
#pragma unroll
    for (int mm = 1; mm < 32; mm <<= 1) vs += __shfl_xor(vs, mm, 32);
    const float rstd = rsqrtf(vs * (1.0f / 512.0f) + 1e-5f);
#pragma unroll
    for (int j = 0; j < 16; ++j) {
        const int c = lane + 32 * j;
        const float oo = (v[j] - mean) * rstd * gw[c] + bw[c];
        if (OUT16) y16[(size_t)row * D_ + c] = (f16)oo;
        else       y32[(size_t)row * D_ + c] = oo;
    }
}

// ---------------- elementwise / prep kernels ----------------
__global__ void cast_f32_f16_k(const float* __restrict__ src, f16* __restrict__ dst,
                               size_t n) {
    size_t i = (size_t)blockIdx.x * blockDim.x + threadIdx.x;
    const size_t st = (size_t)gridDim.x * blockDim.x;
    for (; i < n; i += st) dst[i] = (f16)src[i];
}

__global__ void cast_embedT_k(const float* __restrict__ E, f16* __restrict__ Et) {
    size_t i = (size_t)blockIdx.x * blockDim.x + threadIdx.x;
    const size_t st = (size_t)gridDim.x * blockDim.x, n = (size_t)V_ * C_;
    for (; i < n; i += st) {
        const int v = (int)(i / C_), c = (int)(i % C_);
        Et[(size_t)c * V_ + v] = (f16)E[i];
    }
}

__global__ void cast_conv_w_k(const float* __restrict__ w, f16* __restrict__ dst) {
    size_t i = (size_t)blockIdx.x * blockDim.x + threadIdx.x;
    const size_t st = (size_t)gridDim.x * blockDim.x, n = (size_t)1536 * D_;
    for (; i < n; i += st) {
        const int dout = (int)(i % D_);
        const int k = (int)(i / D_);
        const int din = k % D_, kk = k / D_;
        dst[i] = (f16)w[(size_t)dout * (D_ * 3) + din * 3 + kk];
    }
}

__global__ void im2row_k(const f16* __restrict__ e16, f16* __restrict__ A) {
    size_t i = (size_t)blockIdx.x * blockDim.x + threadIdx.x;
    const size_t st = (size_t)gridDim.x * blockDim.x;
    const size_t n = (size_t)B_ * S_ * 1536;
    for (; i < n; i += st) {
        const int col = (int)(i % 1536);
        const size_t ri = i / 1536;
        const int b = (int)(ri / S_), s = (int)(ri % S_);
        const int kk = col / D_, din = col % D_;
        const int t = 2 * s - 1 + kk;
        A[i] = (t >= 0 && t < T_) ? e16[((size_t)b * T_ + t) * D_ + din] : (f16)0.0f;
    }
}

__global__ void compute_e2_k(const float* __restrict__ E, float* __restrict__ e2) {
    const int v = blockIdx.x * blockDim.x + threadIdx.x;
    if (v >= V_) return;
    float s = 0.0f;
#pragma unroll 8
    for (int c = 0; c < C_; ++c) { const float x = E[(size_t)v * C_ + c]; s += x * x; }
    e2[v] = s;
}

__global__ void init_minbuf_k(unsigned long long* __restrict__ m, int n) {
    const int i = blockIdx.x * blockDim.x + threadIdx.x;
    if (i < n) m[i] = ~0ull;
}

__global__ void vq_gather_k(const f16* __restrict__ E16,
                            const unsigned long long* __restrict__ minbuf,
                            f16* __restrict__ Aq, int M) {
    size_t i = (size_t)blockIdx.x * blockDim.x + threadIdx.x;
    const size_t st = (size_t)gridDim.x * blockDim.x, n = (size_t)M * C_;
    for (; i < n; i += st) {
        const size_t row = i / C_;
        const int c = (int)(i % C_);
        const int idx = (int)(minbuf[row] & 0xffffffffu);
        Aq[i] = E16[(size_t)idx * C_ + c];
    }
}

__global__ void compute_repl_k(const float* __restrict__ E,
                               const float* __restrict__ poutw,
                               const float* __restrict__ poutb,
                               float* __restrict__ repl) {
    const int d = blockIdx.x * blockDim.x + threadIdx.x;
    if (d >= D_) return;
    float s = poutb[d];
#pragma unroll 8
    for (int c = 0; c < C_; ++c)
        s += E[(size_t)(V_ - 1) * C_ + c] * poutw[(size_t)c * D_ + d];
    repl[d] = s;
}

__global__ void upsample_mask_pos_k(const float* __restrict__ q,
                                    const unsigned char* __restrict__ mask,
                                    const float* __restrict__ repl,
                                    const float* __restrict__ pos,
                                    float* __restrict__ x) {
    size_t i = (size_t)blockIdx.x * blockDim.x + threadIdx.x;
    const size_t st = (size_t)gridDim.x * blockDim.x;
    const size_t n = (size_t)B_ * T_ * D_;
    for (; i < n; i += st) {
        const int d = (int)(i % D_);
        const size_t bt = i / D_;
        const int t = (int)(bt % T_), b = (int)(bt / T_);
        const float v = mask[(size_t)b * T_ + t]
                            ? q[((size_t)b * S_ + (t >> 1)) * D_ + d]
                            : repl[d];
        x[i] = v + pos[(size_t)t * D_ + d];
    }
}

__global__ void zero_scalar_k(float* p) { *p = 0.0f; }

__global__ void commit_loss_k(const float* __restrict__ q,
                              const float* __restrict__ x, float* __restrict__ out) {
    __shared__ float red[256];
    const size_t n = (size_t)B_ * S_ * D_;
    size_t i = (size_t)blockIdx.x * blockDim.x + threadIdx.x;
    const size_t st = (size_t)gridDim.x * blockDim.x;
    float s = 0.0f;
    for (; i < n; i += st) { const float d = q[i] - x[i]; s += d * d; }
    red[threadIdx.x] = s;
    __syncthreads();
    for (int w = 128; w > 0; w >>= 1) {
        if (threadIdx.x < w) red[threadIdx.x] += red[threadIdx.x + w];
        __syncthreads();
    }
    if (threadIdx.x == 0) atomicAdd(out, red[0] * (1.0f / (float)n));
}

// =====================================================================
// host launcher
// =====================================================================
static inline int cdiv_i(int a, int b) { return (a + b - 1) / b; }

extern "C" void kernel_launch(void* const* d_in, const int* in_sizes, int n_in,
                              void* d_out, int out_size, void* d_ws, size_t ws_size,
                              hipStream_t stream) {
    (void)in_sizes; (void)n_in; (void)out_size; (void)ws_size;
    const float* embs   = (const float*)d_in[0];
    const unsigned char* mask = (const unsigned char*)d_in[1];   // jnp bool
    const float* conv_w = (const float*)d_in[2];
    const float* conv_b = (const float*)d_in[3];
    const float* mlp_ln_g = (const float*)d_in[4];
    const float* mlp_ln_b = (const float*)d_in[5];
    const float* mlp_w1 = (const float*)d_in[6];
    const float* mlp_b1 = (const float*)d_in[7];
    const float* mlp_w2 = (const float*)d_in[8];
    const float* mlp_b2 = (const float*)d_in[9];
    const float* pin_w  = (const float*)d_in[10];
    const float* pin_b  = (const float*)d_in[11];
    const float* embed  = (const float*)d_in[12];
    const float* pout_w = (const float*)d_in[13];
    const float* pout_b = (const float*)d_in[14];
    const float* pos_emb = (const float*)d_in[15];
    const float* ln1_g = (const float*)d_in[16];
    const float* ln1_b = (const float*)d_in[17];
    const float* wq = (const float*)d_in[18];
    const float* bq = (const float*)d_in[19];
    const float* wk = (const float*)d_in[20];
    const float* bk = (const float*)d_in[21];
    const float* wv = (const float*)d_in[22];
    const float* bv = (const float*)d_in[23];
    const float* wo = (const float*)d_in[24];
    const float* bo = (const float*)d_in[25];
    const float* ln2_g = (const float*)d_in[26];
    const float* ln2_b = (const float*)d_in[27];
    const float* bw1 = (const float*)d_in[28];
    const float* bb1 = (const float*)d_in[29];
    const float* bw2 = (const float*)d_in[30];
    const float* bb2 = (const float*)d_in[31];
    const float* lnp_g = (const float*)d_in[32];
    const float* lnp_b = (const float*)d_in[33];

    float* out_x = (float*)d_out;
    float* out_loss = out_x + (size_t)B_ * T_ * D_;

    char* base = (char*)d_ws;
    size_t off = 0;
    auto alloc = [&](size_t bytes) -> void* {
        void* p = base + off;
        off = (off + bytes + 255) & ~(size_t)255;
        return p;
    };
    const size_t MS = (size_t)B_ * S_;   // 12000
    const size_t MT = (size_t)B_ * T_;   // 24000

    f16* e16    = (f16*)alloc(MT * D_ * 2);
    f16* Aim    = (f16*)alloc(MS * 1536 * 2);
    f16* Wc16   = (f16*)alloc((size_t)1536 * D_ * 2);
    float* xS   = (float*)alloc(MS * D_ * 4);
    f16* xS16   = (f16*)alloc(MS * D_ * 2);
    f16* w1_16  = (f16*)alloc((size_t)D_ * FF_ * 2);
    f16* w2_16  = (f16*)alloc((size_t)FF_ * D_ * 2);
    f16* ff16   = (f16*)alloc(MT * FF_ * 2);
    f16* pin16  = (f16*)alloc((size_t)D_ * C_ * 2);
    f16* Et16   = (f16*)alloc((size_t)C_ * V_ * 2);
    f16* E16    = (f16*)alloc((size_t)V_ * C_ * 2);
    f16* pout16 = (f16*)alloc((size_t)C_ * D_ * 2);
    f16* z16    = (f16*)alloc(MS * C_ * 2);
    unsigned long long* minbuf = (unsigned long long*)alloc(MS * 8);
    f16* Aq16   = (f16*)alloc(MS * C_ * 2);
    float* qproj = (float*)alloc(MS * D_ * 4);
    float* e2   = (float*)alloc((size_t)V_ * 4);
    float* repl = (float*)alloc((size_t)D_ * 4);
    float* x    = (float*)alloc(MT * D_ * 4);
    f16* x16    = (f16*)alloc(MT * D_ * 2);
    f16* q16    = (f16*)alloc(MT * D_ * 2);
    f16* k16    = (f16*)alloc(MT * D_ * 2);
    f16* v16    = (f16*)alloc(MT * D_ * 2);
    f16* o16    = (f16*)alloc(MT * D_ * 2);
    f16* wq16   = (f16*)alloc((size_t)D_ * D_ * 2);
    f16* wk16   = (f16*)alloc((size_t)D_ * D_ * 2);
    f16* wv16   = (f16*)alloc((size_t)D_ * D_ * 2);
    f16* wo16   = (f16*)alloc((size_t)D_ * D_ * 2);

    const dim3 blk256(256), blk32(32);
    const int CAST_GRID = 2048;

    // ---- 1. conv downsample as im2row GEMM (K = 3*512) ----
    cast_f32_f16_k<<<CAST_GRID, blk256, 0, stream>>>(embs, e16, MT * D_);
    cast_conv_w_k<<<CAST_GRID, blk256, 0, stream>>>(conv_w, Wc16);
    im2row_k<<<CAST_GRID, blk256, 0, stream>>>(e16, Aim);
    gemm_wmma_f16<0><<<dim3(cdiv_i((int)MS, 128), D_ / 64), blk256, 0, stream>>>(
        Aim, Wc16, conv_b, nullptr, xS, nullptr, (int)MS, D_, 1536);

    // ---- 2. pre-VQ MLP ----
    layernorm_k<true><<<cdiv_i((int)MS, 8), blk256, 0, stream>>>(
        xS, mlp_ln_g, mlp_ln_b, xS16, nullptr, (int)MS);
    cast_f32_f16_k<<<CAST_GRID, blk256, 0, stream>>>(mlp_w1, w1_16, (size_t)D_ * FF_);
    gemm_wmma_f16<1><<<dim3(cdiv_i((int)MS, 128), FF_ / 64), blk256, 0, stream>>>(
        xS16, w1_16, mlp_b1, nullptr, nullptr, ff16, (int)MS, FF_, D_);
    cast_f32_f16_k<<<CAST_GRID, blk256, 0, stream>>>(mlp_w2, w2_16, (size_t)FF_ * D_);
    gemm_wmma_f16<2><<<dim3(cdiv_i((int)MS, 128), D_ / 64), blk256, 0, stream>>>(
        ff16, w2_16, mlp_b2, xS, xS, nullptr, (int)MS, D_, FF_);

    // ---- 3. VQ ----
    cast_f32_f16_k<<<CAST_GRID, blk256, 0, stream>>>(xS, xS16, MS * D_);
    cast_f32_f16_k<<<CAST_GRID, blk256, 0, stream>>>(pin_w, pin16, (size_t)D_ * C_);
    gemm_wmma_f16<3><<<dim3(cdiv_i((int)MS, 128), C_ / 64), blk256, 0, stream>>>(
        xS16, pin16, pin_b, nullptr, nullptr, z16, (int)MS, C_, D_);
    cast_embedT_k<<<CAST_GRID, blk256, 0, stream>>>(embed, Et16);
    cast_f32_f16_k<<<CAST_GRID, blk256, 0, stream>>>(embed, E16, (size_t)V_ * C_);
    compute_e2_k<<<cdiv_i(V_, 256), blk256, 0, stream>>>(embed, e2);
    init_minbuf_k<<<cdiv_i((int)MS, 256), blk256, 0, stream>>>(minbuf, (int)MS);
    vq_argmin_gemm<<<dim3(cdiv_i((int)MS, 128), cdiv_i(V_, 64)), blk256, 0, stream>>>(
        z16, Et16, e2, minbuf, (int)MS);
    vq_gather_k<<<CAST_GRID, blk256, 0, stream>>>(E16, minbuf, Aq16, (int)MS);
    cast_f32_f16_k<<<CAST_GRID, blk256, 0, stream>>>(pout_w, pout16, (size_t)C_ * D_);
    gemm_wmma_f16<0><<<dim3(cdiv_i((int)MS, 128), D_ / 64), blk256, 0, stream>>>(
        Aq16, pout16, pout_b, nullptr, qproj, nullptr, (int)MS, D_, C_);

    zero_scalar_k<<<1, 1, 0, stream>>>(out_loss);
    commit_loss_k<<<1024, blk256, 0, stream>>>(qproj, xS, out_loss);

    // ---- 4. upsample + mask + pos ----
    compute_repl_k<<<cdiv_i(D_, 256), blk256, 0, stream>>>(embed, pout_w, pout_b, repl);
    upsample_mask_pos_k<<<CAST_GRID, blk256, 0, stream>>>(qproj, mask, repl, pos_emb, x);

    // ---- 5. transformer layers ----
    const float qk_scale = 1.0f * 8.0f / sqrtf((float)DH_);   // == 1.0
    const size_t DD = (size_t)D_ * D_, DF = (size_t)D_ * FF_;
    for (int l = 0; l < L_; ++l) {
        layernorm_k<true><<<cdiv_i((int)MT, 8), blk256, 0, stream>>>(
            x, ln1_g + (size_t)l * D_, ln1_b + (size_t)l * D_, x16, nullptr, (int)MT);
        cast_f32_f16_k<<<CAST_GRID, blk256, 0, stream>>>(wq + l * DD, wq16, DD);
        cast_f32_f16_k<<<CAST_GRID, blk256, 0, stream>>>(wk + l * DD, wk16, DD);
        cast_f32_f16_k<<<CAST_GRID, blk256, 0, stream>>>(wv + l * DD, wv16, DD);
        cast_f32_f16_k<<<CAST_GRID, blk256, 0, stream>>>(wo + l * DD, wo16, DD);
        gemm_wmma_f16<3><<<dim3(cdiv_i((int)MT, 128), D_ / 64), blk256, 0, stream>>>(
            x16, wq16, bq + (size_t)l * D_, nullptr, nullptr, q16, (int)MT, D_, D_);
        gemm_wmma_f16<3><<<dim3(cdiv_i((int)MT, 128), D_ / 64), blk256, 0, stream>>>(
            x16, wk16, bk + (size_t)l * D_, nullptr, nullptr, k16, (int)MT, D_, D_);
        gemm_wmma_f16<3><<<dim3(cdiv_i((int)MT, 128), D_ / 64), blk256, 0, stream>>>(
            x16, wv16, bv + (size_t)l * D_, nullptr, nullptr, v16, (int)MT, D_, D_);
        flash_attention<<<dim3(cdiv_i(T_, 16), H_, B_), blk32, 0, stream>>>(
            q16, k16, v16, o16, qk_scale);
        gemm_wmma_f16<2><<<dim3(cdiv_i((int)MT, 128), D_ / 64), blk256, 0, stream>>>(
            o16, wo16, bo + (size_t)l * D_, x, x, nullptr, (int)MT, D_, D_);
        layernorm_k<true><<<cdiv_i((int)MT, 8), blk256, 0, stream>>>(
            x, ln2_g + (size_t)l * D_, ln2_b + (size_t)l * D_, x16, nullptr, (int)MT);
        cast_f32_f16_k<<<CAST_GRID, blk256, 0, stream>>>(bw1 + l * DF, w1_16, DF);
        gemm_wmma_f16<1><<<dim3(cdiv_i((int)MT, 128), FF_ / 64), blk256, 0, stream>>>(
            x16, w1_16, bb1 + (size_t)l * FF_, nullptr, nullptr, ff16, (int)MT, FF_, D_);
        cast_f32_f16_k<<<CAST_GRID, blk256, 0, stream>>>(bw2 + l * DF, w2_16, DF);
        gemm_wmma_f16<2><<<dim3(cdiv_i((int)MT, 128), D_ / 64), blk256, 0, stream>>>(
            ff16, w2_16, bb2 + (size_t)l * D_, x, x, nullptr, (int)MT, D_, FF_);
    }

    // ---- 6. final LayerNorm -> d_out ----
    layernorm_k<false><<<cdiv_i((int)MT, 8), blk256, 0, stream>>>(
        x, lnp_g, lnp_b, nullptr, out_x, (int)MT);
}
